// SlimYOLOv2_quantize_bnfuse_75282186764865
// MI455X (gfx1250) — matrix operand order repaired and checked by reading, not compiled, not run on hardware
//
#include <hip/hip_runtime.h>
#include <hip/hip_bf16.h>
#include <math.h>
#include <stdint.h>

// ---------------------------------------------------------------------------
// SlimYOLOv2 forward on gfx1250 (MI455X).
// Implicit-GEMM 3x3 conv on v_wmma_f32_16x16x32_f16, f32 accumulate.
// Activations: f16, channels-last, 1-pixel zero halo: [16][H+2][W+2][Cpad].
// GEMM K-order k = (ky*3+kx)*Cinp + ci  => for Cinp%32==0 a 32-wide K-tile is
// 64 contiguous bytes per thread, staged to LDS with
// global_load_async_to_lds_b128 (ASYNCcnt), double-buffered so the next
// tile's global->LDS DMA overlaps the current tile's WMMAs
// (s_wait_asynccnt 4 only drains the oldest in-flight tile).
//   A = weights pre-packed f16 in the exact WMMA A-fragment lane layout
//   2 WMMAs per wave per K-step (wave tile 16x32); block tile 16x256
// ---------------------------------------------------------------------------

typedef __attribute__((ext_vector_type(16))) _Float16 v16h;
typedef __attribute__((ext_vector_type(8)))  _Float16 v8h;
typedef __attribute__((ext_vector_type(8)))  float    v8f;

#define TILE_N 256   // 8 waves * 32
#define BK_PAD 40    // LDS B-tile row stride in halves (80B; rows 16B-aligned)

// ---------------------------------------------------------------------------
// Pack f32 OIHW weights -> f16 A-fragments, K reordered to s*CinP + ci.
// Fragment (mt, kt) = 32 lanes x 16 halves; lane l, element e=2v+p:
//   M = l&15,  K = kt*32 + ((v<4) ? 2v : 16+2(v-4)) + 8*(l>>4) + p
// ---------------------------------------------------------------------------
__global__ void pack_weights_kernel(const float* __restrict__ wgt,
                                    _Float16* __restrict__ out,
                                    int Cout, int Cin, int CinP)
{
    const int K  = CinP * 9;
    const int KT = (K + 31) / 32;
    const int MT = (Cout + 15) / 16;
    const int total = MT * KT * 512;
    for (int i = blockIdx.x * blockDim.x + threadIdx.x; i < total;
         i += gridDim.x * blockDim.x) {
        int e    = i & 15;
        int lane = (i >> 4) & 31;
        int t    = i >> 9;            // mt*KT + kt
        int kt   = t % KT;
        int mt   = t / KT;
        int vv   = e >> 1;
        int par  = e & 1;
        int kb   = ((vv < 4) ? (vv * 2) : (16 + (vv - 4) * 2)) + (lane >> 4) * 8 + par;
        int co   = mt * 16 + (lane & 15);
        int kk   = kt * 32 + kb;      // packed-K: s*CinP + ci
        float v  = 0.f;
        if (co < Cout && kk < K) {
            int s  = kk / CinP;       // filter tap 0..8
            int ci = kk - s * CinP;
            if (ci < Cin)
                v = wgt[(co * Cin + ci) * 9 + s];   // OIHW flat
        }
        out[i] = (_Float16)v;
    }
}

// ---------------------------------------------------------------------------
// Implicit-GEMM conv3x3 SAME stride-1.  In/out: padded channels-last f16.
//   in : [16][H+2][W+2][CINP],  out: [16][H+2][W+2][CPAD]
// ---------------------------------------------------------------------------
template <int CINP, int COUT, int H, int W, int LRELU>
__global__ __launch_bounds__(256) void conv3x3_wmma_kernel(
    const _Float16* __restrict__ in, const v16h* __restrict__ wpk,
    const float* __restrict__ bias, _Float16* __restrict__ out)
{
    constexpr int K    = CINP * 9;
    constexpr int KT   = (K + 31) / 32;
    constexpr int HWp  = H * W;
    constexpr int NTOT = 16 * HWp;                 // B = 16
    constexpr int CPAD = ((COUT + 15) / 16) * 16;
    constexpr int IROW = (W + 2) * CINP;           // strides in halves
    constexpr int IIMG = (H + 2) * IROW;
    constexpr int OROW = (W + 2) * CPAD;
    constexpr int OIMG = (H + 2) * OROW;
    constexpr bool FAST = ((CINP % 32) == 0);

    __shared__ _Float16 Bs[2][TILE_N][BK_PAD];     // double-buffered B tile

    const int tid    = threadIdx.x;
    const int wave   = tid >> 5;
    const int lane   = tid & 31;
    const int laneHi = lane >> 4;
    const int laneLo = lane & 15;
    const int n0     = blockIdx.x * TILE_N;
    const int mt     = blockIdx.y;

    // ---- per-thread fixed im2col column -------------------------------
    int gnf = n0 + tid;
    if constexpr ((NTOT % TILE_N) != 0)
        gnf = (gnf > NTOT - 1) ? NTOT - 1 : gnf;   // clamped cols never stored
    const int bb = gnf / HWp;
    const int rm = gnf - bb * HWp;
    const int y0 = rm / W;
    const int x0 = rm - y0 * W;
    // padded coords: tap (ky,kx) reads padded row y0+ky, col x0+kx
    const _Float16* ibase = in + (size_t)bb * IIMG + (size_t)y0 * IROW
                               + (size_t)x0 * CINP;

    const uint32_t ldsa0 = (uint32_t)(uintptr_t)(&Bs[0][tid][0]);
    const uint32_t ldsa1 = (uint32_t)(uintptr_t)(&Bs[1][tid][0]);

    // issue one 64B/thread K-tile into the LDS buffer at 'la' via async DMA
    auto issue_tile = [&](int kt, uint32_t la) {
        const int s   = (kt * 32) / CINP;          // uniform filter tap
        const int ci0 = (kt * 32) - s * CINP;
        const int ky  = s / 3, kx = s - 3 * ky;
        const uint64_t ga =
            (uint64_t)(uintptr_t)(ibase + ky * IROW + kx * CINP + ci0);
        asm volatile(
            "global_load_async_to_lds_b128 %0, %1, off\n\t"
            "global_load_async_to_lds_b128 %0, %1, off offset:16\n\t"
            "global_load_async_to_lds_b128 %0, %1, off offset:32\n\t"
            "global_load_async_to_lds_b128 %0, %1, off offset:48"
            :: "v"(la), "v"(ga) : "memory");
    };

    // incremental A-fragment pointer (avoids per-iter 64-bit mul/shift chain)
    const v16h* ap = wpk + (size_t)mt * KT * 32 + lane;

    v8f acc0 = {}, acc1 = {};

    if constexpr (FAST)
        issue_tile(0, ldsa0);                      // prologue: preload tile 0

    for (int kt = 0; kt < KT; ++kt) {
        const int cur = kt & 1;

        if constexpr (FAST) {
            if (kt + 1 < KT) {
                issue_tile(kt + 1, cur ? ldsa0 : ldsa1);   // overlap next DMA
                asm volatile("s_wait_asynccnt 4" ::: "memory"); // oldest tile done
            } else {
                asm volatile("s_wait_asynccnt 0" ::: "memory");
            }
        } else {
            // generic path (conv1 CINP=4, conv2 CINP=16): maskless d16 loads
            const int k0 = kt * 32;
            v16h r0, r1;
            #pragma unroll
            for (int k = 0; k < 32; ++k) {
                int  kk  = k0 + k;
                bool kin = ((K % 32) == 0) ? true : (kk < K);
                int  kkc = kin ? kk : 0;
                int  s   = kkc / CINP;
                int  ci  = kkc - s * CINP;
                int  ky  = s / 3, kx = s - 3 * ky;
                _Float16 hv = ibase[ky * IROW + kx * CINP + ci];
                hv = kin ? hv : (_Float16)0.f;
                if (k < 16) r0[k] = hv; else r1[k - 16] = hv;
            }
            *(v16h*)&Bs[cur][tid][0]  = r0;
            *(v16h*)&Bs[cur][tid][16] = r1;
        }
        __syncthreads();

        // ---- A fragment: pre-packed, 32B/lane (2x global_load_b128) ----
        v16h af = *ap;
        ap += 32;
        if (kt + 1 < KT)
            __builtin_prefetch(ap, 0, 3);

        // ---- B fragments: lane -> N = laneLo, K = 8*laneHi+e (e<8), 16+8*laneHi+(e-8) (e>=8) ----
        const int koff = laneHi * 8;
        v16h bf0, bf1;
        #pragma unroll
        for (int e = 0; e < 16; ++e)
            bf0[e] = Bs[cur][wave * 32 + laneLo][koff + (e & 7) + ((e & 8) << 1)];
        #pragma unroll
        for (int e = 0; e < 16; ++e)
            bf1[e] = Bs[cur][wave * 32 + 16 + laneLo][koff + (e & 7) + ((e & 8) << 1)];

        acc0 = __builtin_amdgcn_wmma_f32_16x16x32_f16(false, af, false, bf0,
                                                      (short)0, acc0, false, false);
        acc1 = __builtin_amdgcn_wmma_f32_16x16x32_f16(false, af, false, bf1,
                                                      (short)0, acc1, false, false);
        __syncthreads();                            // all reads done before reuse
    }

    // ---- epilogue: C/D layout N = laneLo, M = r + 8*laneHi --------------
    float bv[8];
    if constexpr ((COUT % 16) == 0) {
        const float4* bp4 = (const float4*)(bias + mt * 16 + laneHi * 8);
        float4 u = bp4[0], w4 = bp4[1];
        bv[0] = u.x;  bv[1] = u.y;  bv[2] = u.z;  bv[3] = u.w;
        bv[4] = w4.x; bv[5] = w4.y; bv[6] = w4.z; bv[7] = w4.w;
    } else {
        #pragma unroll
        for (int r = 0; r < 8; ++r) {
            int co = mt * 16 + laneHi * 8 + r;
            bv[r] = (co < COUT) ? bias[co] : 0.f;
        }
    }

    #pragma unroll
    for (int nsub = 0; nsub < 2; ++nsub) {
        const v8f& acc = nsub ? acc1 : acc0;
        int gn = n0 + wave * 32 + nsub * 16 + laneLo;
        bool nok = ((NTOT % TILE_N) == 0) ? true : (gn < NTOT);
        if (nok) {
            int b   = gn / HWp;
            int rem = gn - b * HWp;
            int y   = rem / W;
            int x   = rem - y * W;
            v8h ov;
            #pragma unroll
            for (int r = 0; r < 8; ++r) {
                float v = acc[r] + bv[r];
                if (LRELU) v = (v > 0.f) ? v : 0.1f * v;
                ov[r] = (_Float16)v;
            }
            // 8 contiguous channels -> one 16B store (CPAD-padded, no guard)
            _Float16* op = out + (size_t)b * OIMG + (size_t)(y + 1) * OROW
                               + (size_t)(x + 1) * CPAD + (mt * 16 + laneHi * 8);
            *(v8h*)op = ov;
        }
    }
}

// ---------------------------------------------------------------------------
// 2x2 stride-2 max pool, padded channels-last f16, vectorized over channels
// ---------------------------------------------------------------------------
template <int C, int H, int W>
__global__ void maxpool2_kernel(const _Float16* __restrict__ in,
                                _Float16* __restrict__ out)
{
    constexpr int HO = H / 2, WO = W / 2;
    constexpr int IROW = (W + 2) * C,  IIMG = (H + 2) * IROW;
    constexpr int OROW = (WO + 2) * C, OIMG = (HO + 2) * OROW;
    constexpr int CV = C / 8;
    constexpr int total = 16 * HO * WO * CV;
    for (int i = blockIdx.x * blockDim.x + threadIdx.x; i < total;
         i += gridDim.x * blockDim.x) {
        int cc = i % CV;
        int t  = i / CV;
        int xo = t % WO; t /= WO;
        int yo = t % HO;
        int b  = t / HO;
        const _Float16* p = in + (size_t)b * IIMG + (size_t)(1 + 2 * yo) * IROW
                               + (size_t)(1 + 2 * xo) * C + cc * 8;
        v8h a0 = *(const v8h*)p;
        v8h a1 = *(const v8h*)(p + C);
        v8h a2 = *(const v8h*)(p + IROW);
        v8h a3 = *(const v8h*)(p + IROW + C);
        v8h m;
        #pragma unroll
        for (int e = 0; e < 8; ++e) {
            _Float16 u = (a0[e] > a1[e]) ? a0[e] : a1[e];
            _Float16 v = (a2[e] > a3[e]) ? a2[e] : a3[e];
            m[e] = (u > v) ? u : v;
        }
        _Float16* op = out + (size_t)b * OIMG + (size_t)(1 + yo) * OROW
                           + (size_t)(1 + xo) * C + cc * 8;
        *(v8h*)op = m;
    }
}

// ---------------------------------------------------------------------------
// Zero the 1-pixel halo of a padded channels-last tensor [16][H+2][W+2][CP]
// ---------------------------------------------------------------------------
template <int CP, int H, int W>
__global__ void halo_zero_kernel(_Float16* __restrict__ buf)
{
    constexpr int ROW  = (W + 2) * CP;
    constexpr int IMG  = (H + 2) * ROW;
    constexpr int RB   = (W + 2) * CP;     // one full border row
    constexpr int perB = 2 * RB + 2 * H * CP;
    constexpr int total = 16 * perB;
    for (int i = blockIdx.x * blockDim.x + threadIdx.x; i < total;
         i += gridDim.x * blockDim.x) {
        int b = i / perB;
        int j = i - b * perB;
        int idx;
        if (j < RB)            idx = j;                          // top row
        else if (j < 2 * RB)   idx = (H + 1) * ROW + (j - RB);   // bottom row
        else {
            int j2 = j - 2 * RB;
            if (j2 < H * CP) { int y = j2 / CP; int c = j2 - y * CP;
                               idx = (1 + y) * ROW + c; }                    // left col
            else             { int j3 = j2 - H * CP; int y = j3 / CP;
                               int c = j3 - y * CP;
                               idx = (1 + y) * ROW + (W + 1) * CP + c; }     // right col
        }
        buf[(size_t)b * IMG + idx] = (_Float16)0.f;
    }
}

// ---------------------------------------------------------------------------
// Input conversion: f32 NCHW [16][3][416][416] -> f16 padded CL [16][418][418][4]
// ---------------------------------------------------------------------------
__global__ void convert_input_kernel(const float* __restrict__ x,
                                     _Float16* __restrict__ out)
{
    constexpr int total = 16 * 416 * 416 * 4;
    for (int i = blockIdx.x * blockDim.x + threadIdx.x; i < total;
         i += gridDim.x * blockDim.x) {
        int c  = i & 3;
        int t  = i >> 2;
        int xx = t % 416; t /= 416;
        int yy = t % 416;
        int b  = t / 416;
        float v = (c < 3) ? x[((b * 3 + c) * 416 + yy) * 416 + xx] : 0.f;
        out[((size_t)b * 418 * 418 + (size_t)(yy + 1) * 418 + (xx + 1)) * 4 + c]
            = (_Float16)v;
    }
}

// ---------------------------------------------------------------------------
// YOLO decode from f16 padded CL pred [16][28][28][128] -> f32 [16][3380][24]
// ---------------------------------------------------------------------------
__global__ void decode_kernel(const _Float16* __restrict__ pred,
                              float* __restrict__ out)
{
    const int WS = 26, HW = 676, A = 5, NC = 20, CP = 128;
    const float anchors[5][2] = {{1.19f, 1.98f}, {2.79f, 4.59f}, {4.53f, 8.92f},
                                 {8.06f, 5.29f}, {10.32f, 10.65f}};
    const int total = 16 * HW * A;
    int tid = blockIdx.x * blockDim.x + threadIdx.x;
    if (tid >= total) return;

    int a  = tid % A;
    int t  = tid / A;
    int hw = t % HW;
    int b  = t / HW;
    int gy = hw / WS;
    int gx = hw - gy * WS;

    const _Float16* pb = pred + ((size_t)b * 28 * 28
                                 + (size_t)(gy + 1) * 28 + (gx + 1)) * CP;

    float conf = (float)pb[a];
    float sig  = 1.f / (1.f + expf(-conf));

    float cls[NC];
    float mx = -1e30f;
    #pragma unroll
    for (int c = 0; c < NC; ++c) {
        cls[c] = (float)pb[A + a * NC + c];
        mx = fmaxf(mx, cls[c]);
    }
    float ssum = 0.f;
    #pragma unroll
    for (int c = 0; c < NC; ++c) {
        cls[c] = expf(cls[c] - mx);
        ssum += cls[c];
    }
    float inv = sig / ssum;

    const int boxc = A * (1 + NC);
    float tx = (float)pb[boxc + a * 4 + 0];
    float ty = (float)pb[boxc + a * 4 + 1];
    float tw = (float)pb[boxc + a * 4 + 2];
    float th = (float)pb[boxc + a * 4 + 3];

    float cx = (1.f / (1.f + expf(-tx)) + (float)gx) * 16.f;
    float cy = (1.f / (1.f + expf(-ty)) + (float)gy) * 16.f;
    float bw = expf(tw) * anchors[a][0] * 16.f;
    float bh = expf(th) * anchors[a][1] * 16.f;

    const float invs = 1.f / 416.f;
    float x1 = fminf(fmaxf((cx - 0.5f * bw) * invs, 0.f), 1.f);
    float y1 = fminf(fmaxf((cy - 0.5f * bh) * invs, 0.f), 1.f);
    float x2 = fminf(fmaxf((cx + 0.5f * bw) * invs, 0.f), 1.f);
    float y2 = fminf(fmaxf((cy + 0.5f * bh) * invs, 0.f), 1.f);

    float* o = out + (long)tid * 24;
    o[0] = x1; o[1] = y1; o[2] = x2; o[3] = y2;
    #pragma unroll
    for (int c = 0; c < NC; ++c) o[4 + c] = cls[c] * inv;
}

// ---------------------------------------------------------------------------
// Host-side orchestration
// ---------------------------------------------------------------------------
static inline int pack_slice_halves(int Cout, int CinP)
{
    int K = CinP * 9;
    return ((Cout + 15) / 16) * ((K + 31) / 32) * 512;
}

static inline void launch_pack(const float* w, _Float16* o, int Cout, int Cin,
                               int CinP, hipStream_t s)
{
    int total = pack_slice_halves(Cout, CinP);
    pack_weights_kernel<<<(total + 255) / 256, 256, 0, s>>>(w, o, Cout, Cin, CinP);
}

template <int CINP, int COUT, int H, int W, int LRELU>
static inline void launch_conv(const _Float16* in, const _Float16* wpk,
                               const float* bias, _Float16* out, hipStream_t s)
{
    constexpr int NTOT = 16 * H * W;
    dim3 grid((NTOT + TILE_N - 1) / TILE_N, (COUT + 15) / 16);
    conv3x3_wmma_kernel<CINP, COUT, H, W, LRELU>
        <<<grid, 256, 0, s>>>(in, (const v16h*)wpk, bias, out);
}

template <int C, int H, int W>
static inline void launch_pool(const _Float16* in, _Float16* out, hipStream_t s)
{
    int total = 16 * (H / 2) * (W / 2) * (C / 8);
    int blocks = (total + 255) / 256;
    if (blocks > 65535) blocks = 65535;
    maxpool2_kernel<C, H, W><<<blocks, 256, 0, s>>>(in, out);
}

template <int CP, int H, int W>
static inline void launch_halo(_Float16* buf, hipStream_t s)
{
    int total = 16 * (2 * (W + 2) * CP + 2 * H * CP);
    int blocks = (total + 255) / 256;
    if (blocks > 65535) blocks = 65535;
    halo_zero_kernel<CP, H, W><<<blocks, 256, 0, s>>>(buf);
}

extern "C" void kernel_launch(void* const* d_in, const int* in_sizes, int n_in,
                              void* d_out, int out_size, void* d_ws, size_t ws_size,
                              hipStream_t stream)
{
    (void)in_sizes; (void)n_in; (void)out_size; (void)ws_size;

    const float* x    = (const float*)d_in[0];
    const float* w1   = (const float*)d_in[1];  const float* b1   = (const float*)d_in[2];
    const float* w2   = (const float*)d_in[3];  const float* b2   = (const float*)d_in[4];
    const float* w3_1 = (const float*)d_in[5];  const float* b3_1 = (const float*)d_in[6];
    const float* w3_2 = (const float*)d_in[7];  const float* b3_2 = (const float*)d_in[8];
    const float* w4_1 = (const float*)d_in[9];  const float* b4_1 = (const float*)d_in[10];
    const float* w4_2 = (const float*)d_in[11]; const float* b4_2 = (const float*)d_in[12];
    const float* w5   = (const float*)d_in[13]; const float* b5   = (const float*)d_in[14];
    const float* w6   = (const float*)d_in[15]; const float* b6   = (const float*)d_in[16];
    const float* w7   = (const float*)d_in[17]; const float* b7   = (const float*)d_in[18];
    const float* wp   = (const float*)d_in[19]; const float* bp   = (const float*)d_in[20];

    // workspace: two f16 ping-pong activation buffers + packed f16 weights
    const size_t MAXH = (size_t)16 * 418 * 418 * 16;   // conv1 out: 44,729,344 halves
    _Float16* buf0 = (_Float16*)d_ws;
    _Float16* buf1 = buf0 + MAXH;
    _Float16* pk   = buf1 + MAXH;                      // 16B-aligned

    _Float16* p1   = pk;                               int o1  = pack_slice_halves( 16,   4);
    _Float16* p2   = p1   + o1;                        int o2  = pack_slice_halves( 32,  16);
    _Float16* p3_1 = p2   + o2;                        int o31 = pack_slice_halves( 64,  32);
    _Float16* p3_2 = p3_1 + o31;                       int o32 = pack_slice_halves( 64,  64);
    _Float16* p4_1 = p3_2 + o32;                       int o41 = pack_slice_halves(128,  64);
    _Float16* p4_2 = p4_1 + o41;                       int o42 = pack_slice_halves(128, 128);
    _Float16* p5   = p4_2 + o42;                       int o5  = pack_slice_halves(256, 128);
    _Float16* p6   = p5   + o5;                        int o6  = pack_slice_halves(256, 256);
    _Float16* p7   = p6   + o6;                        int o7  = pack_slice_halves(256, 256);
    _Float16* pp   = p7   + o7;

    launch_pack(w1,   p1,    16,   3,   4, stream);
    launch_pack(w2,   p2,    32,  16,  16, stream);
    launch_pack(w3_1, p3_1,  64,  32,  32, stream);
    launch_pack(w3_2, p3_2,  64,  64,  64, stream);
    launch_pack(w4_1, p4_1, 128,  64,  64, stream);
    launch_pack(w4_2, p4_2, 128, 128, 128, stream);
    launch_pack(w5,   p5,   256, 128, 128, stream);
    launch_pack(w6,   p6,   256, 256, 256, stream);
    launch_pack(w7,   p7,   256, 256, 256, stream);
    launch_pack(wp,   pp,   125, 256, 256, stream);

    // input f32 NCHW -> f16 padded channels-last
    {
        int total = 16 * 416 * 416 * 4;
        convert_input_kernel<<<(total + 255) / 256, 256, 0, stream>>>(x, buf0);
        launch_halo<4, 416, 416>(buf0, stream);
    }

    launch_conv<  4,  16, 416, 416, 1>(buf0, p1,   b1,   buf1, stream);
    launch_pool< 16, 416, 416>(buf1, buf0, stream);
    launch_halo< 16, 208, 208>(buf0, stream);
    launch_conv< 16,  32, 208, 208, 1>(buf0, p2,   b2,   buf1, stream);
    launch_pool< 32, 208, 208>(buf1, buf0, stream);
    launch_halo< 32, 104, 104>(buf0, stream);
    launch_conv< 32,  64, 104, 104, 1>(buf0, p3_1, b3_1, buf1, stream);
    launch_halo< 64, 104, 104>(buf1, stream);
    launch_conv< 64,  64, 104, 104, 1>(buf1, p3_2, b3_2, buf0, stream);
    launch_pool< 64, 104, 104>(buf0, buf1, stream);
    launch_halo< 64,  52,  52>(buf1, stream);
    launch_conv< 64, 128,  52,  52, 1>(buf1, p4_1, b4_1, buf0, stream);
    launch_halo<128,  52,  52>(buf0, stream);
    launch_conv<128, 128,  52,  52, 1>(buf0, p4_2, b4_2, buf1, stream);
    launch_pool<128,  52,  52>(buf1, buf0, stream);
    launch_halo<128,  26,  26>(buf0, stream);
    launch_conv<128, 256,  26,  26, 1>(buf0, p5,   b5,   buf1, stream);
    launch_halo<256,  26,  26>(buf1, stream);
    launch_conv<256, 256,  26,  26, 1>(buf1, p6,   b6,   buf0, stream);
    launch_halo<256,  26,  26>(buf0, stream);
    launch_conv<256, 256,  26,  26, 1>(buf0, p7,   b7,   buf1, stream);
    launch_halo<256,  26,  26>(buf1, stream);
    launch_conv<256, 125,  26,  26, 0>(buf1, pp,   bp,   buf0, stream);   // pred head

    const int total = 16 * 676 * 5;
    decode_kernel<<<(total + 255) / 256, 256, 0, stream>>>(buf0, (float*)d_out);
}
